// MoDTransformerBlock_1640677507296
// MI455X (gfx1250) — matrix operand order repaired and checked by reading, not compiled, not run on hardware
//
#include <hip/hip_runtime.h>

// ---------------------------------------------------------------------------
// MoD transformer block for gfx1250 (MI455X): f16 WMMA GEMMs, fp32 residual.
// B=8, S=4096, H=1024, NH=16, DH=64, DF=4096, k=512 selected tokens/batch.
// ---------------------------------------------------------------------------

typedef __attribute__((ext_vector_type(16))) _Float16 v16h;
typedef __attribute__((ext_vector_type(8)))  float    v8f;

#define BB 8
#define SS 4096
#define HH 1024
#define NHH 16
#define DHH 64
#define DFF 4096
#define KSEL 512
#define MTOT (BB * KSEL)   // 4096 selected rows total

union FragH { v16h v; float4 f4[2]; _Float16 h[16]; };
union AccF  { v8f  v; float f[8]; };

__device__ __forceinline__ float gelu_tanh(float x) {
  const float k0 = 0.7978845608028654f;   // sqrt(2/pi)
  const float k1 = 0.044715f;
  return 0.5f * x * (1.0f + tanhf(k0 * (x + k1 * x * x * x)));
}

// ---------------------------------------------------------------------------
// Generic single-wave WMMA GEMM: C[64 x 16] tile per wave.
//   C = A[M,K] (row-major, lda)  x  Bt[N,K]^T (Bt row-major, ldb)
// Batched via blockIdx.z with split offsets: off = (z>>zshift)*B1 + (z&msk)*B2
// EPI: 0 = f16 out (+bias if non-null)          (Q/K proj, att*V)
//      1 = f16 out transposed per-head for V:   vt[(b*16+h)*64+d][t]
//      2 = f32 out = (res + c + bias) * scale[m]?  (O-proj, FFN2)
//      3 = f16 out = gelu(c + bias)              (FFN1)
//      4 = f16 out = c * cscale                  (attention scores)
// ---------------------------------------------------------------------------
template<int EPI>
__global__ __launch_bounds__(32)
void wmma_gemm(const _Float16* __restrict__ A, const _Float16* __restrict__ Bt,
               const float* __restrict__ bias, const float* __restrict__ res,
               const float* __restrict__ scale,
               _Float16* __restrict__ out16, float* __restrict__ out32,
               int K, int lda, int ldb, int ldc, float cscale, int zshift,
               unsigned long long aB1, unsigned long long aB2,
               unsigned long long bB1, unsigned long long bB2,
               unsigned long long cB1, unsigned long long cB2) {
  const int lane = threadIdx.x;
  const int lo = lane & 15;
  const int hi = lane >> 4;
  const int n0 = blockIdx.x * 16;
  const int m0 = blockIdx.y * 64;
  const unsigned z = blockIdx.z;
  const unsigned zq = z >> zshift;
  const unsigned zr = z & ((1u << zshift) - 1u);

  const _Float16* Ab = A + (size_t)zq * aB1 + (size_t)zr * aB2;
  const _Float16* Bb = Bt + (size_t)zq * bB1 + (size_t)zr * bB2;
  const size_t cOff = (size_t)zq * cB1 + (size_t)zr * cB2;

  v8f acc0 = {}, acc1 = {}, acc2 = {}, acc3 = {};

  const _Float16* pBrow = Bb + (size_t)(n0 + lo) * ldb + hi * 8;
  const _Float16* pArow = Ab + (size_t)(m0 + lo) * lda + hi * 8;

  for (int k0 = 0; k0 < K; k0 += 32) {
    // B fragment: 32x16, lane = column N; K groups hi*8 and 16+hi*8.
    FragH bf;
    const _Float16* pB = pBrow + k0;
    bf.f4[0] = *(const float4*)(pB);
    bf.f4[1] = *(const float4*)(pB + 16);
    __builtin_prefetch(pB + 64, 0, 1);   // global_prefetch for next tiles

    // A fragments: 16x32 each, lane = row M; K groups hi*8 and 16+hi*8.
    FragH a0, a1, a2, a3;
    const _Float16* pA = pArow + k0;
    a0.f4[0] = *(const float4*)(pA);
    a0.f4[1] = *(const float4*)(pA + 16);
    a1.f4[0] = *(const float4*)(pA + 16 * (size_t)lda);
    a1.f4[1] = *(const float4*)(pA + 16 * (size_t)lda + 16);
    a2.f4[0] = *(const float4*)(pA + 32 * (size_t)lda);
    a2.f4[1] = *(const float4*)(pA + 32 * (size_t)lda + 16);
    a3.f4[0] = *(const float4*)(pA + 48 * (size_t)lda);
    a3.f4[1] = *(const float4*)(pA + 48 * (size_t)lda + 16);

    acc0 = __builtin_amdgcn_wmma_f32_16x16x32_f16(false, a0.v, false, bf.v, (short)0, acc0, false, false);
    acc1 = __builtin_amdgcn_wmma_f32_16x16x32_f16(false, a1.v, false, bf.v, (short)0, acc1, false, false);
    acc2 = __builtin_amdgcn_wmma_f32_16x16x32_f16(false, a2.v, false, bf.v, (short)0, acc2, false, false);
    acc3 = __builtin_amdgcn_wmma_f32_16x16x32_f16(false, a3.v, false, bf.v, (short)0, acc3, false, false);
  }

  v8f accs[4] = {acc0, acc1, acc2, acc3};
  const int n = n0 + lo;
  const float bn = (EPI == 4) ? 0.0f : (bias ? bias[n] : 0.0f);
#pragma unroll
  for (int i = 0; i < 4; ++i) {
    AccF u; u.v = accs[i];
#pragma unroll
    for (int r = 0; r < 8; ++r) {
      const int m = m0 + i * 16 + hi * 8 + r;
      const float c = u.f[r];
      if (EPI == 0) {
        out16[cOff + (size_t)m * ldc + n] = (_Float16)(c + bn);
      } else if (EPI == 1) {
        const int b = m >> 9, t = m & 511;
        const int h = n >> 6, d = n & 63;
        out16[(((size_t)(b * 16 + h) * 64 + d) << 9) + t] = (_Float16)(c + bn);
      } else if (EPI == 2) {
        const size_t o = cOff + (size_t)m * ldc + n;
        float v = res[o] + c + bn;
        if (scale) v *= scale[m];
        out32[o] = v;
      } else if (EPI == 3) {
        out16[cOff + (size_t)m * ldc + n] = (_Float16)gelu_tanh(c + bn);
      } else {
        out16[cOff + (size_t)m * ldc + n] = (_Float16)(c * cscale);
      }
    }
  }
}

// ---------------------------------------------------------------------------
// Passthrough copy of x -> out (float4 vectorized)
// ---------------------------------------------------------------------------
__global__ __launch_bounds__(256)
void copy_kernel(const float4* __restrict__ src, float4* __restrict__ dst, size_t n) {
  const size_t i = (size_t)blockIdx.x * 256 + threadIdx.x;
  if (i < n) dst[i] = src[i];
}

// ---------------------------------------------------------------------------
// Router logits: rw[tok] = dot(x[tok,:], w) + b.  One wave per token.
// ---------------------------------------------------------------------------
__global__ __launch_bounds__(256)
void router_kernel(const float* __restrict__ x, const float* __restrict__ w,
                   const float* __restrict__ b, float* __restrict__ rw) {
  const int wave = threadIdx.x >> 5, lane = threadIdx.x & 31;
  const int tok = blockIdx.x * 8 + wave;
  const float* row = x + (size_t)tok * HH;
  float s = 0.0f;
  for (int i = lane; i < HH; i += 32) s += row[i] * w[i];
  for (int m = 16; m > 0; m >>= 1) s += __shfl_xor(s, m, 32);
  if (lane == 0) rw[tok] = s + b[0];
}

// ---------------------------------------------------------------------------
// Top-k per batch via in-LDS bitonic sort (4096 keys), then sort the top-512
// indices ascending (boolean-mask selection order), emit idx + srw.
// ---------------------------------------------------------------------------
__global__ __launch_bounds__(1024)
void topk_kernel(const float* __restrict__ rw, int* __restrict__ idxOut,
                 float* __restrict__ srwOut) {
  __shared__ float val[SS];
  __shared__ int ind[SS];
  const int b = blockIdx.x, tid = threadIdx.x;
  for (int i = tid; i < SS; i += 1024) { val[i] = rw[b * SS + i]; ind[i] = i; }
  __syncthreads();
  // descending sort by value
  for (int k = 2; k <= SS; k <<= 1)
    for (int j = k >> 1; j > 0; j >>= 1) {
      for (int i = tid; i < SS; i += 1024) {
        const int p = i ^ j;
        if (p > i) {
          const bool up = ((i & k) == 0);
          const float a = val[i], c = val[p];
          if (up ? (a < c) : (a > c)) {
            val[i] = c; val[p] = a;
            const int t = ind[i]; ind[i] = ind[p]; ind[p] = t;
          }
        }
      }
      __syncthreads();
    }
  // ascending sort of the winning 512 indices
  for (int k = 2; k <= KSEL; k <<= 1)
    for (int j = k >> 1; j > 0; j >>= 1) {
      for (int i = tid; i < KSEL; i += 1024) {
        const int p = i ^ j;
        if (p > i) {
          const bool up = ((i & k) == 0);
          const int a = ind[i], c = ind[p];
          if (up ? (a > c) : (a < c)) { ind[i] = c; ind[p] = a; }
        }
      }
      __syncthreads();
    }
  for (int i = tid; i < KSEL; i += 1024) {
    const int t = ind[i];
    idxOut[b * KSEL + i] = t;
    srwOut[b * KSEL + i] = rw[b * SS + t];
  }
}

// ---------------------------------------------------------------------------
// Gather selected rows of x into sel (dense [4096 x 1024] f32)
// ---------------------------------------------------------------------------
__global__ __launch_bounds__(256)
void gather_kernel(const float* __restrict__ x, const int* __restrict__ idx,
                   float* __restrict__ sel) {
  const int r = blockIdx.x;
  const int b = r >> 9;
  const int t = idx[r];
  const float4* src = (const float4*)(x + ((size_t)b * SS + t) * HH);
  float4* dst = (float4*)(sel + (size_t)r * HH);
  dst[threadIdx.x] = src[threadIdx.x];
}

// Scatter y rows back into out at the selected token positions
__global__ __launch_bounds__(256)
void scatter_kernel(const float* __restrict__ y, const int* __restrict__ idx,
                    float* __restrict__ out) {
  const int r = blockIdx.x;
  const int b = r >> 9;
  const int t = idx[r];
  const float4* src = (const float4*)(y + (size_t)r * HH);
  float4* dst = (float4*)(out + ((size_t)b * SS + t) * HH);
  dst[threadIdx.x] = src[threadIdx.x];
}

// ---------------------------------------------------------------------------
// LayerNorm over last dim (1024), f32 in -> f16 out. One 256-thread block/row.
// ---------------------------------------------------------------------------
__global__ __launch_bounds__(256)
void ln_kernel(const float* __restrict__ in, const float* __restrict__ g,
               const float* __restrict__ beta, _Float16* __restrict__ out) {
  __shared__ float s1[256], s2[256];
  const size_t row = blockIdx.x;
  const int tid = threadIdx.x;
  const float4 v = *(const float4*)(in + row * HH + tid * 4);
  s1[tid] = v.x + v.y + v.z + v.w;
  s2[tid] = v.x * v.x + v.y * v.y + v.z * v.z + v.w * v.w;
  __syncthreads();
  for (int off = 128; off > 0; off >>= 1) {
    if (tid < off) { s1[tid] += s1[tid + off]; s2[tid] += s2[tid + off]; }
    __syncthreads();
  }
  const float mean = s1[0] * (1.0f / HH);
  const float var = s2[0] * (1.0f / HH) - mean * mean;
  const float rstd = rsqrtf(var + 1e-5f);
  const int c = tid * 4;
  union { _Float16 h[4]; float2 f2; } o;
  o.h[0] = (_Float16)((v.x - mean) * rstd * g[c + 0] + beta[c + 0]);
  o.h[1] = (_Float16)((v.y - mean) * rstd * g[c + 1] + beta[c + 1]);
  o.h[2] = (_Float16)((v.z - mean) * rstd * g[c + 2] + beta[c + 2]);
  o.h[3] = (_Float16)((v.w - mean) * rstd * g[c + 3] + beta[c + 3]);
  *(float2*)(out + row * HH + c) = o.f2;
}

// ---------------------------------------------------------------------------
// Weight transpose + f32->f16: src [R x C] row-major -> dst [C x R] f16
// ---------------------------------------------------------------------------
__global__ __launch_bounds__(256)
void transpose_f16_kernel(const float* __restrict__ src, _Float16* __restrict__ dst,
                          int R, int C) {
  __shared__ float tile[32][33];
  const int c0 = blockIdx.x * 32, r0 = blockIdx.y * 32;
  for (int dr = threadIdx.y; dr < 32; dr += 8)
    tile[dr][threadIdx.x] = src[(size_t)(r0 + dr) * C + c0 + threadIdx.x];
  __syncthreads();
  for (int dr = threadIdx.y; dr < 32; dr += 8)
    dst[(size_t)(c0 + dr) * R + r0 + threadIdx.x] = (_Float16)tile[threadIdx.x][dr];
}

// ---------------------------------------------------------------------------
// In-place softmax over rows of 512 f16 values. One wave per row.
// ---------------------------------------------------------------------------
__global__ __launch_bounds__(256)
void softmax_kernel(_Float16* __restrict__ s) {
  const int wave = threadIdx.x >> 5, lane = threadIdx.x & 31;
  const int row = blockIdx.x * 8 + wave;
  _Float16* p = s + (size_t)row * 512 + lane * 16;
  union { float4 f4[2]; _Float16 h[16]; } u;
  u.f4[0] = *(const float4*)(p);
  u.f4[1] = *(const float4*)(p + 8);
  float v[16];
  float mx = -1e30f;
#pragma unroll
  for (int i = 0; i < 16; ++i) { v[i] = (float)u.h[i]; mx = fmaxf(mx, v[i]); }
  for (int m = 16; m > 0; m >>= 1) mx = fmaxf(mx, __shfl_xor(mx, m, 32));
  float sum = 0.0f;
#pragma unroll
  for (int i = 0; i < 16; ++i) { v[i] = __expf(v[i] - mx); sum += v[i]; }
  for (int m = 16; m > 0; m >>= 1) sum += __shfl_xor(sum, m, 32);
  const float inv = 1.0f / sum;
#pragma unroll
  for (int i = 0; i < 16; ++i) u.h[i] = (_Float16)(v[i] * inv);
  *(float4*)(p) = u.f4[0];
  *(float4*)(p + 8) = u.f4[1];
}

// ---------------------------------------------------------------------------
// Host-side launch
// ---------------------------------------------------------------------------
static inline char* carve(char*& p, size_t bytes) {
  char* r = p;
  p += (bytes + 255) & ~(size_t)255;
  return r;
}

extern "C" void kernel_launch(void* const* d_in, const int* in_sizes, int n_in,
                              void* d_out, int out_size, void* d_ws, size_t ws_size,
                              hipStream_t stream) {
  (void)in_sizes; (void)n_in; (void)out_size; (void)ws_size;
  const float* x        = (const float*)d_in[0];
  const float* w_router = (const float*)d_in[1];
  const float* b_router = (const float*)d_in[2];
  const float* ln1_g    = (const float*)d_in[3];
  const float* ln1_b    = (const float*)d_in[4];
  const float* wq       = (const float*)d_in[5];
  const float* bq       = (const float*)d_in[6];
  const float* wk       = (const float*)d_in[7];
  const float* bk       = (const float*)d_in[8];
  const float* wv       = (const float*)d_in[9];
  const float* bv       = (const float*)d_in[10];
  const float* wo       = (const float*)d_in[11];
  const float* bo       = (const float*)d_in[12];
  const float* ln2_g    = (const float*)d_in[13];
  const float* ln2_b    = (const float*)d_in[14];
  const float* w1       = (const float*)d_in[15];
  const float* b1       = (const float*)d_in[16];
  const float* w2       = (const float*)d_in[17];
  const float* b2       = (const float*)d_in[18];
  float* out = (float*)d_out;

  char* p = (char*)d_ws;
  float*     rw    = (float*)    carve(p, (size_t)BB * SS * 4);
  int*       idx   = (int*)      carve(p, (size_t)BB * KSEL * 4);
  float*     srw   = (float*)    carve(p, (size_t)BB * KSEL * 4);
  float*     sel   = (float*)    carve(p, (size_t)MTOT * HH * 4);
  _Float16*  h16   = (_Float16*) carve(p, (size_t)MTOT * HH * 2);
  _Float16*  wq_t  = (_Float16*) carve(p, (size_t)HH * HH * 2);
  _Float16*  wk_t  = (_Float16*) carve(p, (size_t)HH * HH * 2);
  _Float16*  wv_t  = (_Float16*) carve(p, (size_t)HH * HH * 2);
  _Float16*  wo_t  = (_Float16*) carve(p, (size_t)HH * HH * 2);
  _Float16*  w1_t  = (_Float16*) carve(p, (size_t)HH * DFF * 2);
  _Float16*  w2_t  = (_Float16*) carve(p, (size_t)DFF * HH * 2);
  _Float16*  q16   = (_Float16*) carve(p, (size_t)MTOT * HH * 2);
  _Float16*  k16   = (_Float16*) carve(p, (size_t)MTOT * HH * 2);
  _Float16*  vt16  = (_Float16*) carve(p, (size_t)MTOT * HH * 2);
  _Float16*  s16   = (_Float16*) carve(p, (size_t)BB * NHH * KSEL * KSEL * 2);
  _Float16*  o16   = (_Float16*) carve(p, (size_t)MTOT * HH * 2);
  float*     x1    = (float*)    carve(p, (size_t)MTOT * HH * 4);
  _Float16*  h2_16 = (_Float16*) carve(p, (size_t)MTOT * HH * 2);
  _Float16*  mid16 = (_Float16*) carve(p, (size_t)MTOT * DFF * 2);
  float*     y     = (float*)    carve(p, (size_t)MTOT * HH * 4);

  // 1) passthrough copy out = x
  {
    const size_t n4 = (size_t)BB * SS * HH / 4;
    copy_kernel<<<dim3((unsigned)((n4 + 255) / 256)), dim3(256), 0, stream>>>(
        (const float4*)x, (float4*)out, n4);
  }
  // 2) router logits
  router_kernel<<<dim3(BB * SS / 8), dim3(256), 0, stream>>>(x, w_router, b_router, rw);
  // 3) per-batch top-k + sorted indices + router weights
  topk_kernel<<<dim3(BB), dim3(1024), 0, stream>>>(rw, idx, srw);
  // 4) gather selected tokens
  gather_kernel<<<dim3(MTOT), dim3(256), 0, stream>>>(x, idx, sel);
  // 5) LN1 -> f16 activations
  ln_kernel<<<dim3(MTOT), dim3(256), 0, stream>>>(sel, ln1_g, ln1_b, h16);
  // 6) weight transposes to f16 [N][K]
  {
    dim3 blk(32, 8);
    transpose_f16_kernel<<<dim3(HH / 32, HH / 32), blk, 0, stream>>>(wq, wq_t, HH, HH);
    transpose_f16_kernel<<<dim3(HH / 32, HH / 32), blk, 0, stream>>>(wk, wk_t, HH, HH);
    transpose_f16_kernel<<<dim3(HH / 32, HH / 32), blk, 0, stream>>>(wv, wv_t, HH, HH);
    transpose_f16_kernel<<<dim3(HH / 32, HH / 32), blk, 0, stream>>>(wo, wo_t, HH, HH);
    transpose_f16_kernel<<<dim3(DFF / 32, HH / 32), blk, 0, stream>>>(w1, w1_t, HH, DFF);
    transpose_f16_kernel<<<dim3(HH / 32, DFF / 32), blk, 0, stream>>>(w2, w2_t, DFF, HH);
  }
  const dim3 wblk(32);
  // 7) Q, K projections (f16 out, +bias)
  wmma_gemm<0><<<dim3(HH / 16, MTOT / 64, 1), wblk, 0, stream>>>(
      h16, wq_t, bq, nullptr, nullptr, q16, nullptr,
      HH, HH, HH, HH, 1.0f, 0, 0, 0, 0, 0, 0, 0);
  wmma_gemm<0><<<dim3(HH / 16, MTOT / 64, 1), wblk, 0, stream>>>(
      h16, wk_t, bk, nullptr, nullptr, k16, nullptr,
      HH, HH, HH, HH, 1.0f, 0, 0, 0, 0, 0, 0, 0);
  // 8) V projection, stored transposed per head: vt[(b*16+h)*64+d][t]
  wmma_gemm<1><<<dim3(HH / 16, MTOT / 64, 1), wblk, 0, stream>>>(
      h16, wv_t, bv, nullptr, nullptr, vt16, nullptr,
      HH, HH, HH, HH, 1.0f, 0, 0, 0, 0, 0, 0, 0);
  // 9) attention scores s = (Q K^T) / 8 per (b, h): M=N=512, K=64
  wmma_gemm<4><<<dim3(KSEL / 16, KSEL / 64, BB * NHH), wblk, 0, stream>>>(
      q16, k16, nullptr, nullptr, nullptr, s16, nullptr,
      DHH, HH, HH, KSEL, 0.125f, 4,
      (unsigned long long)KSEL * HH, (unsigned long long)DHH,
      (unsigned long long)KSEL * HH, (unsigned long long)DHH,
      (unsigned long long)NHH * KSEL * KSEL, (unsigned long long)KSEL * KSEL);
  // 10) softmax in place over rows of 512
  softmax_kernel<<<dim3(BB * NHH * KSEL / 8), dim3(256), 0, stream>>>(s16);
  // 11) O = att * V per (b, h): M=512, N=64, K=512 -> o16[b*512+q][h*64+d]
  wmma_gemm<0><<<dim3(DHH / 16, KSEL / 64, BB * NHH), wblk, 0, stream>>>(
      s16, vt16, nullptr, nullptr, nullptr, o16, nullptr,
      KSEL, KSEL, KSEL, HH, 1.0f, 4,
      (unsigned long long)NHH * KSEL * KSEL, (unsigned long long)KSEL * KSEL,
      (unsigned long long)NHH * DHH * KSEL, (unsigned long long)DHH * KSEL,
      (unsigned long long)KSEL * HH, (unsigned long long)DHH);
  // 12) O projection + residual: x1 = sel + o16 @ wo + bo
  wmma_gemm<2><<<dim3(HH / 16, MTOT / 64, 1), wblk, 0, stream>>>(
      o16, wo_t, bo, sel, nullptr, nullptr, x1,
      HH, HH, HH, HH, 1.0f, 0, 0, 0, 0, 0, 0, 0);
  // 13) LN2 -> f16
  ln_kernel<<<dim3(MTOT), dim3(256), 0, stream>>>(x1, ln2_g, ln2_b, h2_16);
  // 14) FFN1: mid = gelu(h2 @ w1 + b1), f16
  wmma_gemm<3><<<dim3(DFF / 16, MTOT / 64, 1), wblk, 0, stream>>>(
      h2_16, w1_t, b1, nullptr, nullptr, mid16, nullptr,
      HH, HH, HH, DFF, 1.0f, 0, 0, 0, 0, 0, 0, 0);
  // 15) FFN2: y = (x1 + mid @ w2 + b2) * srw[m]
  wmma_gemm<2><<<dim3(HH / 16, MTOT / 64, 1), wblk, 0, stream>>>(
      mid16, w2_t, b2, x1, srw, nullptr, y,
      DFF, DFF, DFF, HH, 1.0f, 0, 0, 0, 0, 0, 0, 0);
  // 16) scatter y back into out at selected positions
  scatter_kernel<<<dim3(MTOT), dim3(256), 0, stream>>>(y, idx, out);
}
